// MultisourcesWindowedCrossAttention_53102975648022
// MI455X (gfx1250) — matrix operand order repaired and checked by reading, not compiled, not run on hardware
//
#include <hip/hip_runtime.h>
#include <hip/hip_bf16.h>
#include <hip/hip_fp16.h>

typedef __attribute__((ext_vector_type(16))) _Float16 v16h;
typedef __attribute__((ext_vector_type(8)))  _Float16 v8h;
typedef __attribute__((ext_vector_type(8)))  float    v8f;

#define WSZ   8
#define NH    8
#define DIM   256
#define INNER 256
#define CIN   64
#define NB    4
#define N1    256     // windows source 1 (16x16)
#define N2    144     // windows source 2 (12x12)
#define NW    400     // total windows
#define NP    416     // padded attention K (13 * 32)
#define QKK   64      // padded qk head dim (32 f + 8 c + pad)

__device__ __forceinline__ v8f wmma_f16(v16h a, v16h b, v8f c) {
  // v_wmma_f32_16x16x32_f16: (neg_a, A, neg_b, B, c_mod, C, reuse_a, reuse_b)
  return __builtin_amdgcn_wmma_f32_16x16x32_f16(false, a, false, b, (short)0, c, false, false);
}

// ---------------------------------------------------------------------------
// Kernel 1: WcT[k][j] = (vb_w @ v_w)[j][k], stored f16 (B-operand ready).
// ---------------------------------------------------------------------------
__global__ void wc_kernel(const float* __restrict__ vb_w,
                          const float* __restrict__ v_w,
                          _Float16* __restrict__ wct) {
  int idx = blockIdx.x * 256 + threadIdx.x;   // idx = k*256 + j
  int k = idx >> 8, j = idx & 255;
  float s = 0.f;
  for (int t = 0; t < 256; ++t) s += vb_w[j * 256 + t] * v_w[t * 256 + k];
  wct[idx] = (_Float16)s;
}

// ---------------------------------------------------------------------------
// Kernel 2: per-window mean -> fqk/cqk projection -> RMS norm -> padded q/k.
// One block (256 thr) per window.
// ---------------------------------------------------------------------------
__global__ void qk_kernel(const float* __restrict__ x1, const float* __restrict__ c1,
                          const float* __restrict__ x2, const float* __restrict__ c2,
                          const float* __restrict__ fqk_w, const float* __restrict__ fqk_b,
                          const float* __restrict__ fqk_g,
                          const float* __restrict__ cqk_w, const float* __restrict__ cqk_b,
                          const float* __restrict__ cqk_g,
                          _Float16* __restrict__ qcat, _Float16* __restrict__ kcat) {
  __shared__ float favg[DIM];
  __shared__ float cavg[4];
  __shared__ float yf[2 * INNER];
  __shared__ float yc[2 * CIN];
  __shared__ float red[256];

  int g = blockIdx.x;
  int b = g / NW, n = g % NW;
  const float* xs; const float* cs; int HW, wh, ww;
  if (n < N1) { xs = x1; cs = c1; HW = 128; wh = n >> 4; ww = n & 15; }
  else { int nn = n - N1; xs = x2; cs = c2; HW = 96; wh = nn / 12; ww = nn % 12; }
  int t = threadIdx.x;

  { // feature mean: thread t = channel t, coalesced across lanes
    float s = 0.f;
    for (int p = 0; p < 64; ++p) {
      int i = p >> 3, j = p & 7;
      s += xs[((size_t)(b * HW + wh * 8 + i) * HW + (ww * 8 + j)) * DIM + t];
    }
    favg[t] = s * (1.0f / 64.0f);
  }
  if (t < 4) {
    float s = 0.f;
    for (int p = 0; p < 64; ++p) {
      int i = p >> 3, j = p & 7;
      s += cs[((size_t)(b * HW + wh * 8 + i) * HW + (ww * 8 + j)) * 4 + t];
    }
    cavg[t] = s * (1.0f / 64.0f);
  }
  __syncthreads();

  { // fqk projection: thread t does output rows t and t+256 (K = 256)
    float a0 = fqk_b[t], a1 = fqk_b[t + INNER];
    const float* w0 = fqk_w + (size_t)t * DIM;
    const float* w1 = fqk_w + (size_t)(t + INNER) * DIM;
    for (int k = 0; k < DIM; ++k) { float f = favg[k]; a0 += f * w0[k]; a1 += f * w1[k]; }
    yf[t] = a0; yf[t + INNER] = a1;
  }
  if (t < 2 * CIN) {
    float a = cqk_b[t];
    const float* w = cqk_w + t * 4;
    for (int k = 0; k < 4; ++k) a += cavg[k] * w[k];
    yc[t] = a;
  }
  __syncthreads();

  // RMS over full 512-wide fqk row
  red[t] = yf[t] * yf[t] + yf[t + INNER] * yf[t + INNER];
  __syncthreads();
  for (int s = 128; s > 0; s >>= 1) { if (t < s) red[t] += red[t + s]; __syncthreads(); }
  float rf = rsqrtf(red[0] * (1.0f / 512.0f) + 1e-6f);
  __syncthreads();
  red[t] = (t < 2 * CIN) ? yc[t] * yc[t] : 0.f;
  __syncthreads();
  for (int s = 128; s > 0; s >>= 1) { if (t < s) red[t] += red[t + s]; __syncthreads(); }
  float rc = rsqrtf(red[0] * (1.0f / 128.0f) + 1e-6f);
  __syncthreads();

  const float qs_f = 0.17677669529663687f;  // 1/sqrt(32)
  const float qs_c = 0.35355339059327373f;  // 1/sqrt(8)
  for (int idx = t; idx < NH * QKK; idx += 256) {
    int h = idx >> 6, kk = idx & 63;
    float qv = 0.f, kv = 0.f;
    if (kk < 32) {
      int o = h * 32 + kk;
      qv = yf[o] * rf * fqk_g[o] * qs_f;
      kv = yf[INNER + o] * rf * fqk_g[INNER + o];
    } else if (kk < 40) {
      int o = h * 8 + (kk - 32);
      qv = yc[o] * rc * cqk_g[o] * qs_c;
      kv = yc[CIN + o] * rc * cqk_g[CIN + o];
    }
    size_t base = ((size_t)(b * NH + h) * NW + n) * QKK + kk;
    qcat[base] = (_Float16)qv;
    kcat[base] = (_Float16)kv;
  }
}

// ---------------------------------------------------------------------------
// Kernel 3: w = x_window @ WcT   (M=102400, K=256, N=256), one wave per
// 16x16 output tile, 8 WMMA K-steps. w stored f16 as [g][p][ch].
// ---------------------------------------------------------------------------
__global__ void wproj_kernel(const float* __restrict__ x1, const float* __restrict__ x2,
                             const _Float16* __restrict__ wct, _Float16* __restrict__ w) {
  int wid = blockIdx.x * 8 + (threadIdx.x >> 5);
  int lane = threadIdx.x & 31;
  int ct = wid & 15;
  int rt = (wid >> 4) & 3;
  int g = wid >> 6;
  int b = g / NW, n = g % NW;
  const float* xs; int HW, wh, ww;
  if (n < N1) { xs = x1; HW = 128; wh = n >> 4; ww = n & 15; }
  else { int nn = n - N1; xs = x2; HW = 96; wh = nn / 12; ww = nn % 12; }

  int row = lane & 15;
  int hi = lane >> 4;                       // 0 or 1
  int p = rt * 16 + row;
  int pi = p >> 3, pj = p & 7;
  const float* xrow = xs + ((size_t)(b * HW + wh * 8 + pi) * HW + (ww * 8 + pj)) * DIM;

  v8f acc = {};
  for (int ks = 0; ks < 8; ++ks) {
    int base = ks * 32;
    int koff = base + hi * 16;
    const float4* pa0 = (const float4*)(xrow + koff);
    const float4* pa1 = (const float4*)(xrow + koff + 8);
    float4 f0 = pa0[0], f1 = pa0[1], f2 = pa1[0], f3 = pa1[1];
    v16h a;
    a[0] = (_Float16)f0.x; a[1] = (_Float16)f0.y; a[2]  = (_Float16)f0.z; a[3]  = (_Float16)f0.w;
    a[4] = (_Float16)f1.x; a[5] = (_Float16)f1.y; a[6]  = (_Float16)f1.z; a[7]  = (_Float16)f1.w;
    a[8] = (_Float16)f2.x; a[9] = (_Float16)f2.y; a[10] = (_Float16)f2.z; a[11] = (_Float16)f2.w;
    a[12] = (_Float16)f3.x; a[13] = (_Float16)f3.y; a[14] = (_Float16)f3.z; a[15] = (_Float16)f3.w;
    v16h bb;
    int kb = base + hi * 16;
    int col = ct * 16 + row;
#pragma unroll
    for (int e = 0; e < 16; ++e) bb[e] = wct[(size_t)(kb + e) * DIM + col];
    acc = wmma_f16(a, bb, acc);
  }
#pragma unroll
  for (int e = 0; e < 8; ++e) {
    int r = e + hi * 8;
    int p_out = rt * 16 + r;
    int ch = ct * 16 + row;
    w[((size_t)g * 64 + p_out) * DIM + ch] = (_Float16)acc[e];
  }
}

// ---------------------------------------------------------------------------
// Kernel 4: scores + mask + softmax -> probs P (f16, padded to 416 cols).
// One block per (b*h, 16-row group): 8 waves split the 25 column tiles,
// then each wave softmaxes 2 rows from LDS.
// ---------------------------------------------------------------------------
__global__ void scores_kernel(const _Float16* __restrict__ qcat,
                              const _Float16* __restrict__ kcat,
                              _Float16* __restrict__ P) {
  __shared__ float S[16][NP];
  int bh = blockIdx.x / 25;
  int rg = blockIdx.x % 25;
  int wv = threadIdx.x >> 5;
  int lane = threadIdx.x & 31;
  int row = lane & 15;
  int hi = lane >> 4;

  int nq = rg * 16 + row;
  const _Float16* qrow = qcat + ((size_t)bh * NW + nq) * QKK;

  for (int ct = wv; ct < 25; ct += 8) {
    int nk = ct * 16 + row;                       // B column = lane&15
    const _Float16* krow = kcat + ((size_t)bh * NW + nk) * QKK;
    v8f acc = {};
#pragma unroll
    for (int ks = 0; ks < 2; ++ks) {
      int base = ks * 32;
      v8h a0 = *(const v8h*)(qrow + base + hi * 16);
      v8h a1 = *(const v8h*)(qrow + base + hi * 16 + 8);
      v8h b0 = *(const v8h*)(krow + base + hi * 16);
      v8h b1 = *(const v8h*)(krow + base + hi * 16 + 8);
      v16h a, bb;
#pragma unroll
      for (int e = 0; e < 8; ++e) { a[e] = a0[e]; a[8 + e] = a1[e]; bb[e] = b0[e]; bb[8 + e] = b1[e]; }
      acc = wmma_f16(a, bb, acc);
    }
#pragma unroll
    for (int e = 0; e < 8; ++e) {
      int r = e + hi * 8;
      int nq2 = rg * 16 + r;
      int m = ct * 16 + row;
      bool allowed = (nq2 < N1) != (m < N1);      // cross-source only
      S[r][m] = allowed ? acc[e] : -3.0e38f;
    }
  }
  __syncthreads();

  for (int rr = 0; rr < 2; ++rr) {
    int r = wv * 2 + rr;
    float mx = -3.4e38f;
    for (int m = lane; m < NW; m += 32) mx = fmaxf(mx, S[r][m]);
    for (int off = 16; off > 0; off >>= 1) mx = fmaxf(mx, __shfl_xor(mx, off, 32));
    float sum = 0.f;
    for (int m = lane; m < NW; m += 32) { float e = __expf(S[r][m] - mx); S[r][m] = e; sum += e; }
    for (int off = 16; off > 0; off >>= 1) sum += __shfl_xor(sum, off, 32);
    float inv = 1.0f / sum;
    _Float16* prow = P + ((size_t)bh * NW + rg * 16 + r) * NP;
    for (int m = lane; m < NP; m += 32)
      prow[m] = (_Float16)((m < NW) ? S[r][m] * inv : 0.f);
  }
}

// ---------------------------------------------------------------------------
// Kernel 5: out = P @ Whead, scattered with un-window mapping.
// Block = (b*h, 16-row tile, group of 128 output cols); LDS-staged A/B.
// ---------------------------------------------------------------------------
__global__ void av_kernel(const _Float16* __restrict__ P,
                          const _Float16* __restrict__ w,
                          float* __restrict__ out) {
  __shared__ _Float16 sA[16][32];
  __shared__ _Float16 sB[32][128];
  int blk = blockIdx.x;
  int ctg = blk & 15;                 // 16 groups of 128 cols = 2048
  int rt = (blk >> 4) % 25;
  int bh = blk / (16 * 25);
  int b = bh >> 3, h = bh & 7;
  int tid = threadIdx.x;
  int wv = tid >> 5, lane = tid & 31;
  int row = lane & 15, hi = lane >> 4;

  v8f acc = {};
  for (int ks = 0; ks < 13; ++ks) {   // K = 416 (tail m>=400 has P==0)
    __syncthreads();
    { // stage A: 16x32 probs tile
      int r = tid >> 4;
      int c2 = (tid & 15) * 2;
      const _Float16* prow = P + ((size_t)bh * NW + rt * 16 + r) * NP + ks * 32 + c2;
      sA[r][c2] = prow[0];
      sA[r][c2 + 1] = prow[1];
    }
    { // stage B: 32x128 slice of Whead, coalesced 16B loads
      int r = tid >> 3;
      int seg = tid & 7;
      int m = ks * 32 + r; if (m > NW - 1) m = NW - 1;   // clamp; P tail is zero
      int c0 = ctg * 128 + seg * 16;
      int pl = c0 >> 8, ch = c0 & 255;
      const _Float16* src = w + (((size_t)(b * NW + m) * 64 + 8 * h + pl) * DIM + ch);
      v8h d0 = ((const v8h*)src)[0], d1 = ((const v8h*)src)[1];
#pragma unroll
      for (int e = 0; e < 8; ++e) { sB[r][seg * 16 + e] = d0[e]; sB[r][seg * 16 + 8 + e] = d1[e]; }
    }
    __syncthreads();
    v16h a, bb;
#pragma unroll
    for (int e = 0; e < 8; ++e) { a[e] = sA[row][hi * 16 + e]; a[8 + e] = sA[row][hi * 16 + 8 + e]; }
    int collocal = wv * 16 + row;
#pragma unroll
    for (int e = 0; e < 16; ++e) bb[e] = sB[hi * 16 + e][collocal];
    acc = wmma_f16(a, bb, acc);
  }

  int ct = ctg * 8 + wv;
  int cg = ct * 16 + row;
  int pl = cg >> 8, ch = cg & 255;    // window position p = 8h + pl -> i = h, j = pl
  int jsp = pl;
#pragma unroll
  for (int e = 0; e < 8; ++e) {
    int r = e + hi * 8;
    int nq = rt * 16 + r;
    size_t off;
    if (nq < N1) {
      int wh = nq >> 4, ww = nq & 15;
      off = ((size_t)(b * 128 + wh * 8 + h) * 128 + (ww * 8 + jsp)) * DIM + ch;
    } else {
      int nn = nq - N1;
      int wh = nn / 12, ww = nn % 12;
      off = (size_t)NB * 128 * 128 * DIM +
            ((size_t)(b * 96 + wh * 8 + h) * 96 + (ww * 8 + jsp)) * DIM + ch;
    }
    out[off] = acc[e];
  }
}

// ---------------------------------------------------------------------------
extern "C" void kernel_launch(void* const* d_in, const int* in_sizes, int n_in,
                              void* d_out, int out_size, void* d_ws, size_t ws_size,
                              hipStream_t stream) {
  const float* x1    = (const float*)d_in[0];
  const float* c1    = (const float*)d_in[1];
  const float* x2    = (const float*)d_in[2];
  const float* c2    = (const float*)d_in[3];
  const float* fqk_w = (const float*)d_in[4];
  const float* fqk_b = (const float*)d_in[5];
  const float* fqk_g = (const float*)d_in[6];
  const float* cqk_w = (const float*)d_in[7];
  const float* cqk_b = (const float*)d_in[8];
  const float* cqk_g = (const float*)d_in[9];
  const float* v_w   = (const float*)d_in[10];
  const float* vb_w  = (const float*)d_in[11];
  float* out = (float*)d_out;

  // workspace layout (bytes)
  char* ws = (char*)d_ws;
  _Float16* wct  = (_Float16*)(ws + 0);          // 256*256*2        = 131072
  _Float16* qcat = (_Float16*)(ws + 131072);     // 32*400*64*2      = 1638400
  _Float16* kcat = (_Float16*)(ws + 1769472);    // 32*400*64*2      = 1638400
  _Float16* P    = (_Float16*)(ws + 3407872);    // 32*400*416*2     = 10649600
  _Float16* w    = (_Float16*)(ws + 14057472);   // 1600*64*256*2    = 52428800

  wc_kernel<<<256, 256, 0, stream>>>(vb_w, v_w, wct);
  qk_kernel<<<NB * NW, 256, 0, stream>>>(x1, c1, x2, c2,
                                         fqk_w, fqk_b, fqk_g,
                                         cqk_w, cqk_b, cqk_g, qcat, kcat);
  wproj_kernel<<<12800, 256, 0, stream>>>(x1, x2, wct, w);       // 102400 tiles / 8 waves
  scores_kernel<<<32 * 25, 256, 0, stream>>>(qcat, kcat, P);
  av_kernel<<<32 * 25 * 16, 256, 0, stream>>>(P, w, out);        // 12800 blocks
}
